// GCNModel_11897059410630
// MI455X (gfx1250) — matrix operand order repaired and checked by reading, not compiled, not run on hardware
//
#include <hip/hip_runtime.h>

// MI455X / gfx1250, wave32.
//  - A@H aggregation GEMMs: v_wmma_f32_16x16x32_bf16, A converted fp32->bf16
//    in-register (HBM-roofline-bound: 3 x 256MB of A ~= 33us at 23.3 TB/s).
//    B tiles staged to LDS with async global->LDS copies (ASYNCcnt pipeline).
//  - Feature GEMMs: v_wmma_f32_16x16x4_f32 (fp32 fidelity, tiny FLOP share).
//  - 51-wide head + pool: fp32 VALU, LDS reduction (deterministic, <2% FLOPs).

typedef __attribute__((ext_vector_type(16))) __bf16 v16bf;
typedef __attribute__((ext_vector_type(8)))  float  v8f;
typedef __attribute__((ext_vector_type(2)))  float  v2f;
typedef __attribute__((ext_vector_type(8)))  float  f8;
typedef __attribute__((ext_vector_type(4)))  unsigned int u32x4;

// Exact parameter types of the async-LDS builtins (from hipcc diagnostics):
// pointee is a GCC-style int4 vector; AS1 (global) src, AS3 (LDS) dst.
typedef int v4i_gcc __attribute__((vector_size(16)));
typedef __attribute__((address_space(1))) v4i_gcc* gas_b128;
typedef __attribute__((address_space(3))) v4i_gcc* las_b128;

#define NNODES 1024
#define NCH    128
#define NBATCH 64
#define NOUT   51

#if __has_builtin(__builtin_amdgcn_global_load_async_to_lds_b128)
#define ASYNC_LDS 1
#else
#define ASYNC_LDS 0
#endif

// 16-byte global -> LDS copy: async (ASYNCcnt-tracked) when available.
__device__ __forceinline__ void copy16_to_lds(__bf16* lds_dst, const __bf16* gsrc) {
#if ASYNC_LDS
  __builtin_amdgcn_global_load_async_to_lds_b128(
      (gas_b128)gsrc, (las_b128)lds_dst, 0, 0);
#else
  *(u32x4*)lds_dst = *(const u32x4*)gsrc;
#endif
}

template <int N>
__device__ __forceinline__ void wait_async() {
#if ASYNC_LDS
#if __has_builtin(__builtin_amdgcn_s_wait_asynccnt)
  __builtin_amdgcn_s_wait_asynccnt(N);
#else
  asm volatile("s_wait_asynccnt %0" ::"i"(N) : "memory");
#endif
#endif
}

// ---------------------------------------------------------------------------
// xw_kernel: Tt[b][c][m] = sum_k In[b][m][k] * W[k][c]     (fp32 math -> bf16)
// Output stored K-major (transposed) so ah_kernel's B-operand loads/copies are
// contiguous. grid.x = NBATCH*(NNODES/16); block = 256 (wave w -> cols w*16..)
// ---------------------------------------------------------------------------
__global__ __launch_bounds__(256) void xw_kernel(
    const float* __restrict__ In, int F,
    const float* __restrict__ W,              // [F][NCH]
    __bf16* __restrict__ Tt)                  // [B][NCH][NNODES]
{
  int b    = blockIdx.x >> 6;
  int m0   = (blockIdx.x & 63) << 4;
  int wave = threadIdx.x >> 5;
  int lane = threadIdx.x & 31;
  int lh   = lane >> 4;        // lane half (0: lanes 0-15, 1: lanes 16-31)
  int lm   = lane & 15;
  int c0   = wave << 4;

  const float* Ib = In + (size_t)b * NNODES * F;
  v8f acc = {};

#if __has_builtin(__builtin_amdgcn_wmma_f32_16x16x4_f32)
  // A 16x4 f32 operand: lane l holds A[l%16][2*(l/16) + v], v = 0,1
  // B 4x16  f32 operand: lane l holds B[2*(l/16) + v][l%16]
  const float* ar = Ib + (size_t)(m0 + lm) * F;
  for (int k0 = 0; k0 < F; k0 += 4) {
    v2f aop = *(const v2f*)(ar + k0 + 2 * lh);
    v2f bop;
    bop[0] = W[(size_t)(k0 + 2 * lh + 0) * NCH + c0 + lm];
    bop[1] = W[(size_t)(k0 + 2 * lh + 1) * NCH + c0 + lm];
    acc = __builtin_amdgcn_wmma_f32_16x16x4_f32(false, aop, false, bop,
                                                (short)0, acc, false, false);
  }
#else
  #pragma unroll
  for (int r = 0; r < 8; r++) {
    int m = m0 + r + 8 * lh, c = c0 + lm;
    float s = 0.f;
    for (int k = 0; k < F; k++)
      s = __builtin_fmaf(Ib[(size_t)m * F + k], W[(size_t)k * NCH + c], s);
    acc[r] = s;
  }
#endif

  // C layout: acc[r] = C[m0 + r + 8*lh][c0 + lm]; store transposed as bf16
  __bf16* Tb = Tt + (size_t)b * NCH * NNODES;
  #pragma unroll
  for (int r = 0; r < 8; r++)
    Tb[(size_t)(c0 + lm) * NNODES + m0 + r + 8 * lh] = (__bf16)acc[r];
}

// ---------------------------------------------------------------------------
// ah_kernel: H[b][n][c] = relu( sum_m A[b][n][m] * Tt[b][c][m] )
// grid.x = NBATCH * 4 (row blocks of 256); block = 256 (wave w -> 32 rows,
// all 128 cols: 16 accumulated 16x16 tiles). Per 32-wide k-step:
//   - B tile (32x128 bf16, 8KB) staged in double-buffered LDS via async
//     global->LDS B128 copies (2 per thread), pipelined one step ahead.
//   - A rows streamed from HBM as fp32 (32B/lane x2 rows), packed-cvt to bf16.
//   - 16x v_wmma_f32_16x16x32_bf16.
// ---------------------------------------------------------------------------
__global__ __launch_bounds__(256) void ah_kernel(
    const float*  __restrict__ A,             // [B][NNODES][NNODES]
    const __bf16* __restrict__ Tt,            // [B][NCH][NNODES]
    float* __restrict__ H)                    // [B][NNODES][NCH]
{
  __shared__ __align__(32) __bf16 sB[2][NCH * 32];   // 2 x 8KB

  int tid  = threadIdx.x;
  int b    = blockIdx.x >> 2;
  int wave = tid >> 5;
  int m0   = ((blockIdx.x & 3) << 8) + (wave << 5);  // 32 rows per wave
  int lane = tid & 31;
  int lh   = lane >> 4;
  int lm   = lane & 15;

  const float*  Ab = A  + (size_t)b * NNODES * NNODES;
  const __bf16* Tb = Tt + (size_t)b * NCH * NNODES;
  const float*  ar0 = Ab + (size_t)(m0 + lm) * NNODES;        // rows 0..15
  const float*  ar1 = Ab + (size_t)(m0 + 16 + lm) * NNODES;   // rows 16..31

  // Staging map: thread copies column ct, k-half hf of each 32x128 B tile.
  int ct = tid >> 1;          // 0..127
  int hf = tid & 1;           // 0,1 -> k offsets 0,16
  const __bf16* gB = Tb + (size_t)ct * NNODES + 16 * hf;      // + k0 later
  int loff = ct * 32 + 16 * hf;

  // Preload tile 0 into buffer 0 (2 async B128 per thread).
  copy16_to_lds(&sB[0][loff],     gB);
  copy16_to_lds(&sB[0][loff + 8], gB + 8);

  v8f acc[2][8] = {};

  for (int k0 = 0; k0 < NNODES; k0 += 32) {
    int cur = (k0 >> 5) & 1;
    if (k0 + 32 < NNODES) {
      __syncthreads();                       // all reads of buf[cur^1] done
      const __bf16* s = gB + k0 + 32;
      copy16_to_lds(&sB[cur ^ 1][loff],     s);
      copy16_to_lds(&sB[cur ^ 1][loff + 8], s + 8);
      wait_async<2>();                       // our tile-k0 copies landed
    } else {
      wait_async<0>();
    }
    __syncthreads();                         // tile k0 visible to all waves

    if (k0 + 128 < NNODES) {
      __builtin_prefetch(ar0 + k0 + 128, 0, 1);   // global_prefetch_b8
      __builtin_prefetch(ar1 + k0 + 128, 0, 1);
    }

    // A 16x32 bf16 operand layout: lane l holds M=l%16,
    //   elems [0..7]  -> K = k0 + 16*(l/16) + j
    //   elems [8..15] -> K = k0 + 16*(l/16) + 8 + j
    f8 a00 = *(const f8*)(ar0 + k0 + 16 * lh);
    f8 a01 = *(const f8*)(ar0 + k0 + 16 * lh + 8);
    f8 a10 = *(const f8*)(ar1 + k0 + 16 * lh);
    f8 a11 = *(const f8*)(ar1 + k0 + 16 * lh + 8);
    v16bf aop0, aop1;
    #pragma unroll
    for (int j = 0; j < 8; j++) {
      aop0[j] = (__bf16)a00[j];  aop0[8 + j] = (__bf16)a01[j];
      aop1[j] = (__bf16)a10[j];  aop1[8 + j] = (__bf16)a11[j];
    }

    // B 32x16 bf16 operand: lane l holds N=l%16, K = 16*(l/16) + [0..16)
    #pragma unroll
    for (int nt = 0; nt < 8; nt++) {
      v16bf bop = *(const v16bf*)&sB[cur][((nt << 4) + lm) * 32 + 16 * lh];
      acc[0][nt] = __builtin_amdgcn_wmma_f32_16x16x32_bf16(
          false, aop0, false, bop, (short)0, acc[0][nt], false, false);
      acc[1][nt] = __builtin_amdgcn_wmma_f32_16x16x32_bf16(
          false, aop1, false, bop, (short)0, acc[1][nt], false, false);
    }
  }

  // ReLU epilogue; C layout: acc[g][nt][r] = C[m0+16g + r + 8*lh][nt*16 + lm]
  float* Hb = H + (size_t)b * NNODES * NCH;
  #pragma unroll
  for (int g = 0; g < 2; g++) {
    #pragma unroll
    for (int nt = 0; nt < 8; nt++) {
      #pragma unroll
      for (int r = 0; r < 8; r++) {
        float v = acc[g][nt][r];
        v = v > 0.f ? v : 0.f;
        Hb[(size_t)(m0 + 16 * g + r + 8 * lh) * NCH + (nt << 4) + lm] = v;
      }
    }
  }
}

// ---------------------------------------------------------------------------
// dense_pool_kernel: out[b][o] = sum_n relu( sum_c H[b][n][c] * Wd[c][o] )
// <2% of total FLOPs -> fp32 VALU; LDS weight cache + LDS tree reduction.
// ---------------------------------------------------------------------------
__global__ __launch_bounds__(1024) void dense_pool_kernel(
    const float* __restrict__ H,              // [B][NNODES][NCH]
    const float* __restrict__ Wd,             // [NCH][NOUT]
    float* __restrict__ out)                  // [B][NOUT]
{
  __shared__ float sW[NCH * NOUT];
  __shared__ float red[16][64];
  int b = blockIdx.x;
  for (int i = threadIdx.x; i < NCH * NOUT; i += 1024) sW[i] = Wd[i];
  __syncthreads();

  int o  = threadIdx.x & 63;
  int ng = threadIdx.x >> 6;                  // 0..15: node-strip group
  const float* Hb = H + (size_t)b * NNODES * NCH;

  float s = 0.f;
  if (o < NOUT) {
    for (int n = ng; n < NNODES; n += 16) {
      const float* h = Hb + (size_t)n * NCH;
      float d = 0.f;
      #pragma unroll 4
      for (int c = 0; c < NCH; c++) d = __builtin_fmaf(h[c], sW[c * NOUT + o], d);
      s += d > 0.f ? d : 0.f;
    }
  }
  red[ng][o] = s;
  __syncthreads();
  if (ng == 0 && o < NOUT) {
    float t = 0.f;
    #pragma unroll
    for (int g = 0; g < 16; g++) t += red[g][o];
    out[(size_t)b * NOUT + o] = t;
  }
}

// ---------------------------------------------------------------------------
extern "C" void kernel_launch(void* const* d_in, const int* in_sizes, int n_in,
                              void* d_out, int out_size, void* d_ws, size_t ws_size,
                              hipStream_t stream) {
  (void)in_sizes; (void)n_in; (void)out_size; (void)ws_size;
  const float* x  = (const float*)d_in[0];   // [64][1024][64]
  const float* a  = (const float*)d_in[1];   // [64][1024][1024]
  const float* W0 = (const float*)d_in[2];   // [64][128]
  const float* W1 = (const float*)d_in[3];   // [128][128]
  const float* W2 = (const float*)d_in[4];   // [128][128]
  const float* Wd = (const float*)d_in[5];   // [128][51]
  float* out = (float*)d_out;                // [64][51]

  // Workspace: Tt (bf16, 16MB) then H (fp32, 32MB) = 48MB total.
  char* ws = (char*)d_ws;
  __bf16* Tt = (__bf16*)ws;
  float*  H  = (float*)(ws + (size_t)NBATCH * NCH * NNODES * sizeof(__bf16));

  const dim3 blk(256);
  const dim3 gXW(NBATCH * (NNODES / 16));    // 4096 blocks
  const dim3 gAH(NBATCH * 4);                // 256 blocks, 2048 waves

  // Layer 0: T = X @ W0 ; H = relu(A @ T)
  xw_kernel<<<gXW, blk, 0, stream>>>(x, 64,  W0, Tt);
  ah_kernel<<<gAH, blk, 0, stream>>>(a, Tt, H);
  // Layer 1
  xw_kernel<<<gXW, blk, 0, stream>>>(H, 128, W1, Tt);
  ah_kernel<<<gAH, blk, 0, stream>>>(a, Tt, H);
  // Layer 2
  xw_kernel<<<gXW, blk, 0, stream>>>(H, 128, W2, Tt);
  ah_kernel<<<gAH, blk, 0, stream>>>(a, Tt, H);
  // Head: relu(H @ Wd) summed over nodes
  dense_pool_kernel<<<dim3(NBATCH), dim3(1024), 0, stream>>>(H, Wd, out);
}
